// SetToGraphGNN_17351667876297
// MI455X (gfx1250) — matrix operand order repaired and checked by reading, not compiled, not run on hardware
//
#include <hip/hip_runtime.h>

typedef __attribute__((ext_vector_type(16))) _Float16 v16h;
typedef __attribute__((ext_vector_type(8)))  _Float16 v8h;
typedef __attribute__((ext_vector_type(8)))  float    v8f;

#define NNODES 16384      // B*N = 16*1024
#define NEDGES 163840     // NNODES*10

// ---- WMMA operand loaders (wave32, 16x16x32 f16 layouts per CDNA5 ISA 7.12.2) ----
// A (16x32): lane<16 holds row=lane, halves {K+0..7, K+16..23}; lane>=16 same rows,
// halves {K+8..15, K+24..31}. Caller pre-offsets pointer by (lane>>4)*8.
__device__ __forceinline__ v16h ldA(const _Float16* __restrict__ p) {
  v8h lo = *(const v8h*)p;
  v8h hi = *(const v8h*)(p + 16);
  v16h r;
#pragma unroll
  for (int i = 0; i < 8; i++) { r[i] = lo[i]; r[i + 8] = hi[i]; }
  return r;
}
// B (32x16): lane holds column=lane%16, halves {K+0..7, K+16..23} (lane<16) / {K+8..15, K+24..31}; caller pre-offsets by (lane>>4)*8.
__device__ __forceinline__ v16h ldB(const _Float16* __restrict__ p) {
  v8h lo = *(const v8h*)p;
  v8h hi = *(const v8h*)(p + 16);
  v16h r;
#pragma unroll
  for (int i = 0; i < 8; i++) { r[i] = lo[i]; r[i + 8] = hi[i]; }
  return r;
}

// ---------------- Stage 1: kNN (top-10 nearest, excluding self) ----------------
// grid = 64 (16 batches x 4 row-chunks), block = 256. One thread per row.
__global__ void __launch_bounds__(256) knn_kernel(const float* __restrict__ x,
                                                  int* __restrict__ nbr) {
  __shared__ float xs[256 * 16];
  __shared__ float sqs[256];
  const int b = blockIdx.x >> 2;
  const int n = ((blockIdx.x & 3) << 8) + threadIdx.x;
  const float* xb = x + (size_t)b * 1024 * 16;

  float xr[16];
#pragma unroll
  for (int c = 0; c < 16; c++) xr[c] = xb[n * 16 + c];
  float sqn = 0.f;
#pragma unroll
  for (int c = 0; c < 16; c++) sqn += xr[c] * xr[c];

  float bd[10]; int bi[10];
#pragma unroll
  for (int j = 0; j < 10; j++) { bd[j] = 3.4e38f; bi[j] = 0; }

  for (int m0 = 0; m0 < 1024; m0 += 256) {
    __syncthreads();
#pragma unroll
    for (int t = 0; t < 16; t++)
      xs[t * 256 + threadIdx.x] = xb[m0 * 16 + t * 256 + threadIdx.x];
    __syncthreads();
    {
      float s = 0.f;
#pragma unroll
      for (int c = 0; c < 16; c++) { float v = xs[threadIdx.x * 16 + c]; s += v * v; }
      sqs[threadIdx.x] = s;
    }
    __syncthreads();
    for (int mi = 0; mi < 256; mi++) {
      const int m = m0 + mi;
      float dot = 0.f;
#pragma unroll
      for (int c = 0; c < 16; c++) dot += xr[c] * xs[mi * 16 + c];
      const float d2 = sqn + sqs[mi] - 2.f * dot;
      if (m != n && d2 < bd[9]) {
        float pd = d2; int pi = m;                 // stable insert (ascending m => top_k tie order)
#pragma unroll
        for (int j = 0; j < 10; j++) {
          if (pd < bd[j]) { float td = bd[j]; int ti = bi[j]; bd[j] = pd; bi[j] = pi; pd = td; pi = ti; }
        }
      }
    }
  }
  const int base = (b * 1024 + n) * 10;
#pragma unroll
  for (int j = 0; j < 10; j++) nbr[base + j] = b * 1024 + bi[j];
}

// ---------------- helpers ----------------
__global__ void pack_x_kernel(const float* __restrict__ x, float* __restrict__ h) {
  const int i = blockIdx.x * 256 + threadIdx.x;   // < NNODES*32
  const int nrow = i >> 5, c = i & 31;
  h[i] = (c < 16) ? x[(nrow << 4) + c] : 0.f;
}

__global__ void fill0_kernel(float* __restrict__ p) {
  p[(size_t)blockIdx.x * 256 + threadIdx.x] = 0.f;
}

// edge-parallel scatter-add: agg[dst] += h[src]; grid = NEDGES*D/256
__global__ void scatter_kernel(const float* __restrict__ h, const int* __restrict__ nbr,
                               float* __restrict__ agg, int dshift) {
  const int gid = blockIdx.x * 256 + threadIdx.x;
  const int e = gid >> dshift;
  const int c = gid & ((1 << dshift) - 1);
  const int srcn = (int)((unsigned)e / 10u);
  const int dstn = nbr[e];
  const float v = h[((size_t)srcn << dshift) + c];
  __hip_atomic_fetch_add(&agg[((size_t)dstn << dshift) + c], v,
                         __ATOMIC_RELAXED, __HIP_MEMORY_SCOPE_AGENT);
}

// f32 -> f16 with K padding (cols >= in become 0); total = rows << ishift
__global__ void cvtpad_kernel(const float* __restrict__ s, _Float16* __restrict__ d,
                              int in, int ishift, int total) {
  const int i = blockIdx.x * 256 + threadIdx.x;
  if (i >= total) return;
  const int r = i >> ishift;
  const int c = i & ((1 << ishift) - 1);
  d[i] = (c < in) ? (_Float16)s[r * in + c] : (_Float16)0.f;
}

// ---------------- dual-GEMM: out = relu?(A1@B1^T + A2@B2^T + bias) ----------------
// One wave computes a 16x64 strip (4 col-tiles): A fragments loaded once per K-step
// feed 8 WMMAs -> ~2.5x higher arithmetic intensity vs 1 tile/wave (we are L2-BW bound).
__global__ void __launch_bounds__(256) gemm_dual4_kernel(
    const _Float16* __restrict__ A1, const _Float16* __restrict__ B1,
    const _Float16* __restrict__ A2, const _Float16* __restrict__ B2,
    const float* __restrict__ bias, float* __restrict__ out,
    int K, int Dout, int colGroups, int relu) {
  const int tileId = blockIdx.x * 8 + (threadIdx.x >> 5);
  const int rowTile = tileId / colGroups;
  const int colGroup = tileId - rowTile * colGroups;
  const int lane = threadIdx.x & 31;
  const int lh = lane & 15, hi = lane >> 4;
  const int arow = (rowTile << 4) + lh;
  const int ocol = (colGroup << 6) + lh;          // first of 4 col-tiles (stride 16)
  const _Float16* a1 = A1 + (size_t)arow * K + (hi << 3);
  const _Float16* a2 = A2 + (size_t)arow * K + (hi << 3);
  const _Float16* b1 = B1 + (size_t)ocol * K + (hi << 3);
  const _Float16* b2 = B2 + (size_t)ocol * K + (hi << 3);
  const size_t tstride = (size_t)16 * K;          // next col-tile = +16 rows of W

  v8f acc[4];
#pragma unroll
  for (int t = 0; t < 4; t++) {
    const float bv = bias[ocol + 16 * t];
#pragma unroll
    for (int r = 0; r < 8; r++) acc[t][r] = bv;
  }

  for (int k0 = 0; k0 < K; k0 += 32) {
    const v16h a1v = ldA(a1 + k0);
    const v16h a2v = ldA(a2 + k0);
#pragma unroll
    for (int t = 0; t < 4; t++) {
      acc[t] = __builtin_amdgcn_wmma_f32_16x16x32_f16(false, a1v, false,
                 ldB(b1 + t * tstride + k0), (short)0, acc[t], false, false);
      acc[t] = __builtin_amdgcn_wmma_f32_16x16x32_f16(false, a2v, false,
                 ldB(b2 + t * tstride + k0), (short)0, acc[t], false, false);
    }
  }

  const int orow = (rowTile << 4) + (hi << 3);
#pragma unroll
  for (int t = 0; t < 4; t++) {
#pragma unroll
    for (int r = 0; r < 8; r++) {
      float v = acc[t][r];
      if (relu) v = fmaxf(v, 0.f);
      out[(size_t)(orow + r) * Dout + ocol + 16 * t] = v;
    }
  }
}

// ---------------- gram: out[b,n,m] = sum_d H[b,n,d]*H[b,m,d], H = 1024x64 ----------------
// Same 4-wide col blocking; 16 col-groups per batch, 64 row-tiles per batch.
__global__ void __launch_bounds__(256) gram4_kernel(const _Float16* __restrict__ h16,
                                                    float* __restrict__ out) {
  const int tileId = blockIdx.x * 8 + (threadIdx.x >> 5);
  const int b = tileId >> 10;                     // 64*16 = 1024 tiles per batch
  const int t = tileId & 1023;
  const int rowTile = t >> 4, colGroup = t & 15;
  const int lane = threadIdx.x & 31;
  const int lh = lane & 15, hi = lane >> 4;
  const _Float16* H = h16 + ((size_t)b << 16);    // 1024*64 halves per batch
  const _Float16* ap = H + (size_t)((rowTile << 4) + lh) * 64 + (hi << 3);
  const _Float16* bp = H + (size_t)((colGroup << 6) + lh) * 64 + (hi << 3);

  v8f acc[4];
#pragma unroll
  for (int tt = 0; tt < 4; tt++)
#pragma unroll
    for (int r = 0; r < 8; r++) acc[tt][r] = 0.f;

#pragma unroll
  for (int k0 = 0; k0 < 64; k0 += 32) {
    const v16h av = ldA(ap + k0);
#pragma unroll
    for (int tt = 0; tt < 4; tt++) {
      acc[tt] = __builtin_amdgcn_wmma_f32_16x16x32_f16(false, av, false,
                  ldB(bp + (size_t)tt * 16 * 64 + k0), (short)0, acc[tt], false, false);
    }
  }

  const size_t ob = ((size_t)b << 20);
  const int orow = (rowTile << 4) + (hi << 3);
  const int oc0 = (colGroup << 6) + lh;
#pragma unroll
  for (int tt = 0; tt < 4; tt++)
#pragma unroll
    for (int r = 0; r < 8; r++)
      out[ob + (size_t)(orow + r) * 1024 + oc0 + 16 * tt] = acc[tt][r];
}

// ---------------- workspace layout (all offsets 256B-aligned; ~65 MB total) ----------------
#define OFF_NBR   ((size_t)0)          // int[NNODES*10]            655,360 B
#define OFF_HA    ((size_t)655360)     // f32 NNODES*256         16,777,216 B
#define OFF_HB    ((size_t)17432576)   // f32 NNODES*256         16,777,216 B
#define OFF_AGG   ((size_t)34209792)   // f32 NNODES*256         16,777,216 B
#define OFF_H16   ((size_t)50987008)   // f16 NNODES*256          8,388,608 B
#define OFF_A16   ((size_t)59375616)   // f16 NNODES*256          8,388,608 B
#define OFF_WR16  ((size_t)67764224)   // f16 256*256               131,072 B
#define OFF_WE16  ((size_t)67895296)   // f16 256*256               131,072 B

extern "C" void kernel_launch(void* const* d_in, const int* in_sizes, int n_in,
                              void* d_out, int out_size, void* d_ws, size_t ws_size,
                              hipStream_t stream) {
  (void)in_sizes; (void)n_in; (void)out_size; (void)ws_size;
  const float* x = (const float*)d_in[0];
  const float* Wr[4] = {(const float*)d_in[1], (const float*)d_in[4],
                        (const float*)d_in[7], (const float*)d_in[10]};
  const float* We[4] = {(const float*)d_in[2], (const float*)d_in[5],
                        (const float*)d_in[8], (const float*)d_in[11]};
  const float* bb[4] = {(const float*)d_in[3], (const float*)d_in[6],
                        (const float*)d_in[9], (const float*)d_in[12]};

  char* ws = (char*)d_ws;
  int*      nbr   = (int*)(ws + OFF_NBR);
  float*    hA    = (float*)(ws + OFF_HA);
  float*    hB    = (float*)(ws + OFF_HB);
  float*    agg   = (float*)(ws + OFF_AGG);
  _Float16* h16   = (_Float16*)(ws + OFF_H16);
  _Float16* agg16 = (_Float16*)(ws + OFF_A16);
  _Float16* wr16  = (_Float16*)(ws + OFF_WR16);
  _Float16* we16  = (_Float16*)(ws + OFF_WE16);

  // 1) build kNN edge list
  knn_kernel<<<64, 256, 0, stream>>>(x, nbr);

  // 2) h0 = x, padded to K=32
  pack_x_kernel<<<(NNODES * 32) / 256, 256, 0, stream>>>(x, hA);

  const int ins[4]     = {16, 128, 256, 256};   // true input dims
  const int inShift[4] = {5, 7, 8, 8};          // log2 of padded input dims {32,128,256,256}
  const int outs[4]    = {128, 256, 256, 64};

  float* hc = hA;
  float* hn = hB;
  for (int l = 0; l < 4; l++) {
    const int ish = inShift[l];
    const int inp = 1 << ish;
    const int out = outs[l];
    const int nEl = NNODES * inp;

    fill0_kernel<<<nEl / 256, 256, 0, stream>>>(agg);
    scatter_kernel<<<(NEDGES * inp) / 256, 256, 0, stream>>>(hc, nbr, agg, ish);

    cvtpad_kernel<<<nEl / 256, 256, 0, stream>>>(hc, h16, inp, ish, nEl);      // h already padded
    cvtpad_kernel<<<nEl / 256, 256, 0, stream>>>(agg, agg16, inp, ish, nEl);   // agg already padded
    const int wEl = out * inp;
    cvtpad_kernel<<<wEl / 256, 256, 0, stream>>>(Wr[l], wr16, ins[l], ish, wEl);
    cvtpad_kernel<<<wEl / 256, 256, 0, stream>>>(We[l], we16, ins[l], ish, wEl);

    const int colGroups = out / 64;               // 4 col-tiles (64 cols) per wave
    const int tiles = (NNODES / 16) * colGroups;
    gemm_dual4_kernel<<<tiles / 8, 256, 0, stream>>>(agg16, we16, h16, wr16, bb[l], hn,
                                                     inp, out, colGroups, (l < 3) ? 1 : 0);
    float* tswap = hc; hc = hn; hn = tswap;
  }

  // 3) gram matrices: convert final h (NNODES x 64) to f16, then H H^T per batch
  cvtpad_kernel<<<(NNODES * 64) / 256, 256, 0, stream>>>(hc, h16, 64, 6, NNODES * 64);
  // 16 batches * 1024 tiles = 16384 tiles, 8 waves/block -> 2048 blocks
  gram4_kernel<<<2048, 256, 0, stream>>>(h16, (float*)d_out);
}
